// Attention_18940805775470
// MI455X (gfx1250) — matrix operand order repaired and checked by reading, not compiled, not run on hardware
//
#include <hip/hip_runtime.h>
#include <hip/hip_bf16.h>
#include <math.h>

typedef __attribute__((ext_vector_type(16))) _Float16 v16h;
typedef __attribute__((ext_vector_type(8)))  _Float16 v8h;
typedef __attribute__((ext_vector_type(8)))  float    v8f;

constexpr int   BDIM = 8;       // batch
constexpr int   CDIM = 256;     // channels
constexpr int   NHH  = 8;       // heads
constexpr int   KDD  = 16;      // key dim per head
constexpr int   HDD  = 32;      // value dim per head
constexpr int   NPIX = 1024;    // H*W
constexpr int   HOUT = 512;     // qkv output channels
constexpr float ATT_SCALE = 0.25f;  // KDD^-0.5
constexpr float BN_EPS    = 1e-3f;

// ---------------------------------------------------------------------------
// BN fold: scale = g*rsqrt(v+eps), bias = b - m*scale. q rows also get
// ATT_SCALE folded (softmax logits are q^T k * scale).
// ---------------------------------------------------------------------------
__global__ void k_prep(const float* qg, const float* qb, const float* qm, const float* qv,
                       const float* pg, const float* pb, const float* pm, const float* pv,
                       const float* rg, const float* rb, const float* rm, const float* rv,
                       float* qkv_sc, float* qkv_bi, float* pe_sc, float* pe_bi,
                       float* pr_sc, float* pr_bi) {
  int o = threadIdx.x;                       // 512 threads
  if (o < HOUT) {
    float inv = qg[o] * rsqrtf(qv[o] + BN_EPS);
    float bi  = qb[o] - qm[o] * inv;
    if ((o & 63) < KDD) { inv *= ATT_SCALE; bi *= ATT_SCALE; }   // q rows
    qkv_sc[o] = inv; qkv_bi[o] = bi;
  }
  if (o < CDIM) {
    float i1 = pg[o] * rsqrtf(pv[o] + BN_EPS);
    pe_sc[o] = i1; pe_bi[o] = pb[o] - pm[o] * i1;
    float i2 = rg[o] * rsqrtf(rv[o] + BN_EPS);
    pr_sc[o] = i2; pr_bi[o] = rb[o] - rm[o] * i2;
  }
}

__global__ void k_cvt_w(const float* __restrict__ w, const float* __restrict__ sc,
                        _Float16* __restrict__ wf) {
  int i = blockIdx.x * 256 + threadIdx.x;    // rows of 256 cols, scale per row
  wf[i] = (_Float16)(w[i] * sc[i >> 8]);
}

// x [b][c][n] f32 -> xT [b][n][c] f16 (coalesced read, one-time scatter write)
__global__ void k_cvt_xT(const float* __restrict__ x, _Float16* __restrict__ xT) {
  int i = blockIdx.x * 256 + threadIdx.x;    // 8*256*1024 elements
  int n = i & 1023, c = (i >> 10) & 255, b = i >> 18;
  xT[((size_t)b * NPIX + n) * CDIM + c] = (_Float16)x[i];
}

// ---------------------------------------------------------------------------
// QKV GEMM: one wave per 16x64 block of (512 x 1024) per batch. K=256 in 8
// steps; the A (weight) fragment is loaded once per step and reused by 4
// B fragments / 4 accumulators (32 WMMAs per wave). Epilogue routes rows to
// qT/kT (transposed, [bh][n][kd]) or vF ([bh][hd][n]) with BN bias added.
// ---------------------------------------------------------------------------
__global__ __launch_bounds__(128) void k_gemm_qkv(
    const _Float16* __restrict__ wf,   // [512][256] scale-folded
    const _Float16* __restrict__ xT,   // [b][1024][256]
    const float*    __restrict__ bias, // [512]
    _Float16* __restrict__ qT, _Float16* __restrict__ kT, _Float16* __restrict__ vF) {
  int wave = threadIdx.x >> 5, lane = threadIdx.x & 31;
  int hi = lane >> 4, l15 = lane & 15;
  int t = blockIdx.x * 4 + wave;             // 4096 waves: b(8) x to(32) x tn4(16)
  int b = t >> 9, rem = t & 511;
  int to = rem >> 4, tn4 = rem & 15;
  int o0 = to * 16, n0 = tn4 * 64;

  v8f acc[4];
  for (int j = 0; j < 4; ++j)
    for (int i = 0; i < 8; ++i) acc[j][i] = 0.f;
  const _Float16* wrow = wf + (size_t)(o0 + l15) * CDIM;
  const _Float16* xrow = xT + ((size_t)b * NPIX + n0 + l15) * CDIM;
  for (int cb = 0; cb < CDIM; cb += 32) {
    v8h w0 = *reinterpret_cast<const v8h*>(wrow + cb + hi * 16);
    v8h w1 = *reinterpret_cast<const v8h*>(wrow + cb + hi * 16 + 8);
    v16h a; for (int e = 0; e < 8; ++e) { a[e] = w0[e]; a[8 + e] = w1[e]; }
#pragma unroll
    for (int j = 0; j < 4; ++j) {
      v16h bx = *reinterpret_cast<const v16h*>(xrow + (size_t)j * 16 * CDIM + cb + hi * 16);
      acc[j] = __builtin_amdgcn_wmma_f32_16x16x32_f16(false, a, false, bx, (short)0,
                                                      acc[j], false, false);
    }
  }
  int seg = (o0 >> 4) & 3;   // 0:q 1:k 2:v[0..15] 3:v[16..31]
  int head = o0 >> 6;
  int hi8 = hi * 8;
#pragma unroll
  for (int j = 0; j < 4; ++j) {
    int n = n0 + j * 16 + l15;
    if (seg < 2) {
      v8h outv;
      for (int r = 0; r < 8; ++r) outv[r] = (_Float16)(acc[j][r] + bias[o0 + hi8 + r]);
      _Float16* dst = (seg == 0) ? qT : kT;
      *reinterpret_cast<v8h*>(dst + ((size_t)(b * NHH + head) * NPIX + n) * KDD + hi8) = outv;
    } else {
      int d0 = ((seg == 2) ? 0 : 16) + hi8;
      _Float16* vb = vF + (size_t)(b * NHH + head) * HDD * NPIX;
      for (int r = 0; r < 8; ++r)
        vb[(size_t)(d0 + r) * NPIX + n] = (_Float16)(acc[j][r] + bias[o0 + hi8 + r]);
    }
  }
}

// ---------------------------------------------------------------------------
// Flash attention: one wave per (bh, 16-query block). 32 keys/iter:
// 2 S-WMMAs (K=16 zero-padded to 32) + online softmax + P via LDS -> 2 PV-WMMAs.
// Next key block's K/V lines are prefetched (global_prefetch_b8) one iteration
// ahead to hide L2/HBM latency.
// ---------------------------------------------------------------------------
__global__ __launch_bounds__(128) void k_attn(
    const _Float16* __restrict__ qT, const _Float16* __restrict__ kT,
    const _Float16* __restrict__ vF, float* __restrict__ aout) {
  __shared__ __align__(16) _Float16 lds[4][16][40];
  int wave = threadIdx.x >> 5, lane = threadIdx.x & 31;
  int hi = lane >> 4, l15 = lane & 15;
  int bh = blockIdx.x;                       // 0..63
  int qbase = (blockIdx.y * 4 + wave) * 16;  // query block base

  v16h aq; for (int e = 0; e < 16; ++e) aq[e] = (_Float16)0.f;
  if (hi == 0) {   // lanes 0..15 carry k=0..15; lanes 16..31 (k>=16) stay zero (pad)
    aq = *reinterpret_cast<const v16h*>(
        qT + ((size_t)bh * NPIX + qbase + l15) * KDD);
  }
  float mrow[8], lrow[8];
  v8f o1, o2, zero;
  for (int i = 0; i < 8; ++i) { o1[i] = 0.f; o2[i] = 0.f; zero[i] = 0.f;
                                mrow[i] = -1e30f; lrow[i] = 0.f; }
  const _Float16* kb = kT + (size_t)bh * NPIX * KDD;
  const _Float16* vb = vF + (size_t)bh * HDD * NPIX;

  for (int m0 = 0; m0 < NPIX; m0 += 32) {
    {   // prefetch next key block (wrapped so addresses stay in-buffer)
      int mp = (m0 + 32) & (NPIX - 1);
      __builtin_prefetch(kb + (size_t)(mp + l15) * KDD, 0, 0);
      __builtin_prefetch(vb + (size_t)l15 * NPIX + mp, 0, 0);
      __builtin_prefetch(vb + (size_t)(16 + l15) * NPIX + mp, 0, 0);
    }
    v16h bk0, bk1;
    if (hi == 0) {   // lanes 0..15 carry k(d)=0..15; lanes 16..31 are K-pad -> 0
      bk0 = *reinterpret_cast<const v16h*>(kb + (size_t)(m0 + l15) * KDD);
      bk1 = *reinterpret_cast<const v16h*>(kb + (size_t)(m0 + 16 + l15) * KDD);
    } else {
      for (int e = 0; e < 16; ++e) { bk0[e] = (_Float16)0.f; bk1[e] = (_Float16)0.f; }
    }
    v8f s0 = __builtin_amdgcn_wmma_f32_16x16x32_f16(false, aq, false, bk0, (short)0,
                                                    zero, false, false);
    v8f s1 = __builtin_amdgcn_wmma_f32_16x16x32_f16(false, aq, false, bk1, (short)0,
                                                    zero, false, false);
    float p0[8], p1[8];
    for (int r = 0; r < 8; ++r) {
      float rm = fmaxf(s0[r], s1[r]);
      rm = fmaxf(rm, __shfl_xor(rm, 1, 32));
      rm = fmaxf(rm, __shfl_xor(rm, 2, 32));
      rm = fmaxf(rm, __shfl_xor(rm, 4, 32));
      rm = fmaxf(rm, __shfl_xor(rm, 8, 32));
      float mn = fmaxf(mrow[r], rm);
      float sc = __expf(mrow[r] - mn);
      mrow[r] = mn;
      p0[r] = __expf(s0[r] - mn);
      p1[r] = __expf(s1[r] - mn);
      float su = p0[r] + p1[r];
      su += __shfl_xor(su, 1, 32); su += __shfl_xor(su, 2, 32);
      su += __shfl_xor(su, 4, 32); su += __shfl_xor(su, 8, 32);
      lrow[r] = lrow[r] * sc + su;
      o1[r] *= sc; o2[r] *= sc;
    }
    __syncthreads();
    for (int r = 0; r < 8; ++r) {              // C-layout -> LDS tile (16x32)
      int row = r + 8 * hi;
      lds[wave][row][l15]      = (_Float16)p0[r];
      lds[wave][row][16 + l15] = (_Float16)p1[r];
    }
    __syncthreads();
    int cb0 = hi * 16;                         // A-layout gather from LDS
    v8h pa = *reinterpret_cast<const v8h*>(&lds[wave][l15][cb0]);
    v8h pbv = *reinterpret_cast<const v8h*>(&lds[wave][l15][cb0 + 8]);
    v16h ap; for (int e = 0; e < 8; ++e) { ap[e] = pa[e]; ap[8 + e] = pbv[e]; }
    int kbase = hi * 16;                       // B-fragment keys per lane half
    v16h bv0 = *reinterpret_cast<const v16h*>(vb + (size_t)l15 * NPIX + m0 + kbase);
    v16h bv1 = *reinterpret_cast<const v16h*>(vb + (size_t)(16 + l15) * NPIX + m0 + kbase);
    o1 = __builtin_amdgcn_wmma_f32_16x16x32_f16(false, ap, false, bv0, (short)0, o1,
                                                false, false);
    o2 = __builtin_amdgcn_wmma_f32_16x16x32_f16(false, ap, false, bv1, (short)0, o2,
                                                false, false);
  }
  float* ob = aout + (size_t)bh * HDD * NPIX;  // aout[b][c=h*32+d][n]
  int row0 = qbase + 8 * hi;
  for (int r = 0; r < 8; ++r) {
    float inv = 1.0f / lrow[r];
    ob[(size_t)l15 * NPIX + row0 + r]        = o1[r] * inv;
    ob[(size_t)(16 + l15) * NPIX + row0 + r] = o2[r] * inv;
  }
}

// ---------------------------------------------------------------------------
// Depthwise 3x3 + BN on v, add attention out, emit transposed f16 activation.
// ---------------------------------------------------------------------------
__global__ void k_pe(const _Float16* __restrict__ vF, const float* __restrict__ aout,
                     const float* __restrict__ pw, const float* __restrict__ pe_sc,
                     const float* __restrict__ pe_bi, _Float16* __restrict__ xaT) {
  int bc = blockIdx.x;                 // 0..2047 : b*256 + c
  int b = bc >> 8, c = bc & 255;
  const _Float16* vp = vF + (size_t)bc * NPIX;   // vF channel order == c
  float w9[9];
  for (int i = 0; i < 9; ++i) w9[i] = pw[c * 9 + i];
  float sc = pe_sc[c], bi = pe_bi[c];
  for (int p = threadIdx.x; p < NPIX; p += 256) {
    int y = p >> 5, x = p & 31;
    float s = 0.f;
    for (int ky = 0; ky < 3; ++ky) {
      int yy = y + ky - 1; if (yy < 0 || yy > 31) continue;
      for (int kx = 0; kx < 3; ++kx) {
        int xx = x + kx - 1; if (xx < 0 || xx > 31) continue;
        s += (float)vp[yy * 32 + xx] * w9[ky * 3 + kx];
      }
    }
    float val = s * sc + bi + aout[(size_t)bc * NPIX + p];
    xaT[((size_t)b * NPIX + p) * CDIM + c] = (_Float16)val;
  }
}

// ---------------------------------------------------------------------------
// Projection GEMM: (256x256)@(256x1024) per batch, 16x64 block per wave,
// f32 output + BN bias.
// ---------------------------------------------------------------------------
__global__ __launch_bounds__(128) void k_gemm_proj(
    const _Float16* __restrict__ wf,   // [256][256] scale-folded
    const _Float16* __restrict__ xaT,  // [b][1024][256]
    const float*    __restrict__ bias, // [256]
    float* __restrict__ out) {
  int wave = threadIdx.x >> 5, lane = threadIdx.x & 31;
  int hi = lane >> 4, l15 = lane & 15;
  int t = blockIdx.x * 4 + wave;             // 2048 waves: b(8) x to(16) x tn4(16)
  int b = t >> 8, rem = t & 255;
  int to = rem >> 4, tn4 = rem & 15;
  int o0 = to * 16, n0 = tn4 * 64;
  v8f acc[4];
  for (int j = 0; j < 4; ++j)
    for (int i = 0; i < 8; ++i) acc[j][i] = 0.f;
  const _Float16* wrow = wf + (size_t)(o0 + l15) * CDIM;
  const _Float16* xrow = xaT + ((size_t)b * NPIX + n0 + l15) * CDIM;
  for (int cb = 0; cb < CDIM; cb += 32) {
    v8h w0 = *reinterpret_cast<const v8h*>(wrow + cb + hi * 16);
    v8h w1 = *reinterpret_cast<const v8h*>(wrow + cb + hi * 16 + 8);
    v16h a; for (int e = 0; e < 8; ++e) { a[e] = w0[e]; a[8 + e] = w1[e]; }
#pragma unroll
    for (int j = 0; j < 4; ++j) {
      v16h bx = *reinterpret_cast<const v16h*>(xrow + (size_t)j * 16 * CDIM + cb + hi * 16);
      acc[j] = __builtin_amdgcn_wmma_f32_16x16x32_f16(false, a, false, bx, (short)0,
                                                      acc[j], false, false);
    }
  }
  int hi8 = hi * 8;
#pragma unroll
  for (int j = 0; j < 4; ++j) {
    int n = n0 + j * 16 + l15;
    for (int r = 0; r < 8; ++r) {
      int o = o0 + hi8 + r;
      out[((size_t)b * CDIM + o) * NPIX + n] = acc[j][r] + bias[o];
    }
  }
}

// ---------------------------------------------------------------------------
extern "C" void kernel_launch(void* const* d_in, const int* in_sizes, int n_in,
                              void* d_out, int out_size, void* d_ws, size_t ws_size,
                              hipStream_t stream) {
  const float* x      = (const float*)d_in[0];
  const float* qkv_w  = (const float*)d_in[1];
  const float* qkv_g  = (const float*)d_in[2];
  const float* qkv_b  = (const float*)d_in[3];
  const float* qkv_m  = (const float*)d_in[4];
  const float* qkv_v  = (const float*)d_in[5];
  const float* pe_w   = (const float*)d_in[6];
  const float* pe_g   = (const float*)d_in[7];
  const float* pe_b   = (const float*)d_in[8];
  const float* pe_m   = (const float*)d_in[9];
  const float* pe_v   = (const float*)d_in[10];
  const float* proj_w = (const float*)d_in[11];
  const float* proj_g = (const float*)d_in[12];
  const float* proj_b = (const float*)d_in[13];
  const float* proj_m = (const float*)d_in[14];
  const float* proj_v = (const float*)d_in[15];

  char* ws = (char*)d_ws;
  size_t off = 0;
  auto alloc = [&](size_t bytes) -> void* {
    off = (off + 255) & ~(size_t)255;
    void* p = ws + off;
    off += bytes;
    return p;
  };
  float* qkv_sc = (float*)alloc(HOUT * 4);
  float* qkv_bi = (float*)alloc(HOUT * 4);
  float* pe_sc  = (float*)alloc(CDIM * 4);
  float* pe_bi  = (float*)alloc(CDIM * 4);
  float* pr_sc  = (float*)alloc(CDIM * 4);
  float* pr_bi  = (float*)alloc(CDIM * 4);
  _Float16* wq  = (_Float16*)alloc((size_t)HOUT * CDIM * 2);
  _Float16* wp  = (_Float16*)alloc((size_t)CDIM * CDIM * 2);
  _Float16* xT  = (_Float16*)alloc((size_t)BDIM * NPIX * CDIM * 2);
  _Float16* qTb = (_Float16*)alloc((size_t)BDIM * NHH * NPIX * KDD * 2);
  _Float16* kTb = (_Float16*)alloc((size_t)BDIM * NHH * NPIX * KDD * 2);
  _Float16* vFb = (_Float16*)alloc((size_t)BDIM * NHH * HDD * NPIX * 2);
  float* aout   = (float*)alloc((size_t)BDIM * CDIM * NPIX * 4);
  _Float16* xaT = (_Float16*)alloc((size_t)BDIM * NPIX * CDIM * 2);

  k_prep<<<1, 512, 0, stream>>>(qkv_g, qkv_b, qkv_m, qkv_v,
                                pe_g, pe_b, pe_m, pe_v,
                                proj_g, proj_b, proj_m, proj_v,
                                qkv_sc, qkv_bi, pe_sc, pe_bi, pr_sc, pr_bi);
  k_cvt_w<<<HOUT, 256, 0, stream>>>(qkv_w, qkv_sc, wq);
  k_cvt_w<<<CDIM, 256, 0, stream>>>(proj_w, pr_sc, wp);
  k_cvt_xT<<<(BDIM * CDIM * NPIX) / 256, 256, 0, stream>>>(x, xT);
  k_gemm_qkv<<<1024, 128, 0, stream>>>(wq, xT, qkv_bi, qTb, kTb, vFb);
  k_attn<<<dim3(BDIM * NHH, 16), 128, 0, stream>>>(qTb, kTb, vFb, aout);
  k_pe<<<BDIM * CDIM, 256, 0, stream>>>(vFb, aout, pe_w, pe_sc, pe_bi, xaT);
  k_gemm_proj<<<512, 128, 0, stream>>>(wp, xaT, pr_bi, (float*)d_out);
}
